// DiscriminatorV2_79199196938401
// MI455X (gfx1250) — matrix operand order repaired and checked by reading, not compiled, not run on hardware
//
#include <hip/hip_runtime.h>
#include <stdint.h>

// ---------------------------------------------------------------------------
// DiscriminatorV2 forward for MI455X (gfx1250, wave32, WMMA bf16).
// Pipeline:
//   x  = leaky(LN(traj@w_mlp+b)) + emb[label]          (fp32 math, bf16 out)
//   q  = x@wq+bq ; k = h@wk+bk ; vT = (h@wv+bv)^T      (WMMA bf16, f32 acc,
//                                                       4 out-tiles per wave)
//   attn: per 16-row q tile, scores in LDS, softmax, P@V (WMMA, bf16 probs,
//         valid_len-bounded), row-scaled epilogue, fused max-pool (atomic max)
//   out = sigmoid((pooled + le) @ w_out + b_out)
// ---------------------------------------------------------------------------

#define DD 256   // feature dim
#define BB 64    // batch
#define TT 512   // query length
#define NN 512   // key length

typedef __bf16 bf16_t;
typedef bf16_t v16bf __attribute__((ext_vector_type(16)));
typedef float  v8f   __attribute__((ext_vector_type(8)));

union ABFrag { v16bf v; uint32_t u[8]; };
union CDFrag { v8f v; float f[8]; };

__device__ __forceinline__ uint16_t f2bf(float f) {   // round-to-nearest-even
  uint32_t u = __float_as_uint(f);
  u += 0x7FFFu + ((u >> 16) & 1u);
  return (uint16_t)(u >> 16);
}

// A-matrix 16x32 bf16 fragment (ISA 7.12.2): lane l & l+16 hold row M=l.
// VGPR i : K = hi*16 + 2i,2i+1 (contiguous, matches B fragment)
__device__ __forceinline__ void load_a_frag(ABFrag& a, const uint16_t* __restrict__ row,
                                            int kbase, int hi) {
  const int kb = kbase + hi * 16;
#pragma unroll
  for (int i = 0; i < 8; ++i) {
    const int k = kb + 2 * i;
    a.u[i] = *(const uint32_t*)(row + k);
  }
}

// B-matrix 32x16 bf16 fragment: lane (l, hi) holds col N=l, K = hi*16 + 0..15
// contiguous -> 32B contiguous per lane from `col` + off.
__device__ __forceinline__ void load_b_frag(ABFrag& b, const uint16_t* __restrict__ col,
                                            int off) {
#pragma unroll
  for (int i = 0; i < 8; ++i) b.u[i] = *(const uint32_t*)(col + off + 2 * i);
}

__device__ __forceinline__ v8f wmma_bf16(const ABFrag& a, const ABFrag& b, v8f c) {
  return __builtin_amdgcn_wmma_f32_16x16x32_bf16(false, a.v, false, b.v,
                                                 (short)0, c, false, false);
}

// --------------------------- prep kernels ----------------------------------

__global__ __launch_bounds__(256) void k_convert_bf16(const float* __restrict__ src,
                                                      uint16_t* __restrict__ dst, int n) {
  const int i = blockIdx.x * blockDim.x + threadIdx.x;
  if (i < n) dst[i] = f2bf(src[i]);
}

// wT[o*D + i] = w[i*D + o]  (bf16) so GEMM B-fragments read contiguous K.
__global__ __launch_bounds__(256) void k_transpose_w(const float* __restrict__ src,
                                                     uint16_t* __restrict__ dst) {
  const int idx = blockIdx.x * blockDim.x + threadIdx.x;  // D*D
  const int o = idx >> 8, i = idx & 255;
  dst[o * DD + i] = f2bf(src[i * DD + o]);
}

__global__ __launch_bounds__(256) void k_init_pooled(uint32_t* __restrict__ p, int n) {
  const int i = blockIdx.x * blockDim.x + threadIdx.x;
  if (i < n) p[i] = 0xFF800000u;  // -inf
}

// x = leaky_relu(LayerNorm(traj @ w_mlp + b_mlp)) + emb[labels]  -> bf16
__global__ __launch_bounds__(256) void k_compute_x(const float* __restrict__ traj,
    const int* __restrict__ labels, const float* __restrict__ emb,
    const float* __restrict__ w_mlp, const float* __restrict__ b_mlp,
    const float* __restrict__ ln_g, const float* __restrict__ ln_b,
    uint16_t* __restrict__ xb) {
  const int d = threadIdx.x;                 // 0..255
  const int t = blockIdx.x, b = blockIdx.y;
  const float s0 = traj[((size_t)b * TT + t) * 2 + 0];
  const float s1 = traj[((size_t)b * TT + t) * 2 + 1];
  const float y = s0 * w_mlp[d] + s1 * w_mlp[DD + d] + b_mlp[d];
  __shared__ float sh1[8], sh2[8];
  float v1 = y, v2 = y * y;
#pragma unroll
  for (int m = 1; m < 32; m <<= 1) {
    v1 += __shfl_xor(v1, m, 32);
    v2 += __shfl_xor(v2, m, 32);
  }
  if ((d & 31) == 0) { sh1[d >> 5] = v1; sh2[d >> 5] = v2; }
  __syncthreads();
  if (d == 0) {
    float t1 = 0.f, t2 = 0.f;
    for (int i = 0; i < 8; ++i) { t1 += sh1[i]; t2 += sh2[i]; }
    sh1[0] = t1; sh2[0] = t2;
  }
  __syncthreads();
  const float mean = sh1[0] * (1.0f / 256.0f);
  const float var  = sh2[0] * (1.0f / 256.0f) - mean * mean;
  float val = (y - mean) * rsqrtf(var + 1e-5f) * ln_g[d] + ln_b[d];
  val = (val > 0.0f) ? val : 0.01f * val;                // leaky_relu
  val += emb[labels[b] * DD + d];
  xb[((size_t)b * TT + t) * DD + d] = f2bf(val);
}

// --------------------------- projection GEMM -------------------------------
// out[b, m, :] = in[b, m, :] @ wT^T + bias.
// 4 waves/block (M dim); each wave computes 4 consecutive 16x16 output tiles,
// reusing every A fragment 4x (quarter the activation traffic, 32 WMMA/wave).
__global__ __launch_bounds__(128) void k_gemm_proj(const uint16_t* __restrict__ in,
    const uint16_t* __restrict__ wT, const float* __restrict__ bias,
    uint16_t* __restrict__ out, int M, int transposed) {
  const int lane = threadIdx.x & 31;
  const int warp = threadIdx.x >> 5;
  const int l = lane & 15, hi = lane >> 4;
  const int mt  = blockIdx.x * 4 + warp;  // M tile
  const int nt0 = blockIdx.y * 4;         // first of 4 out-dim tiles
  const int b   = blockIdx.z;
  const uint16_t* arow = in + ((size_t)b * M + mt * 16 + l) * DD;
  const uint16_t* bcol = wT + (size_t)(nt0 * 16 + l) * DD;
  CDFrag acc[4];
#pragma unroll
  for (int j = 0; j < 4; ++j)
#pragma unroll
    for (int i = 0; i < 8; ++i) acc[j].f[i] = 0.0f;
#pragma unroll
  for (int kc = 0; kc < 8; ++kc) {        // K = 256 in 8 chunks of 32
    ABFrag a;
    load_a_frag(a, arow, kc * 32, hi);
#pragma unroll
    for (int j = 0; j < 4; ++j) {
      ABFrag bb;
      load_b_frag(bb, bcol + (size_t)j * 16 * DD, kc * 32 + hi * 16);
      acc[j].v = wmma_bf16(a, bb, acc[j].v);
    }
  }
  const int mb = mt * 16 + hi * 8;        // C/D: vgpr r -> row r + hi*8, col = l
#pragma unroll
  for (int j = 0; j < 4; ++j) {
    const int col = (nt0 + j) * 16 + l;
    const float bvv = bias[col];
    if (!transposed) {
#pragma unroll
      for (int r = 0; r < 8; ++r)
        out[((size_t)b * M + mb + r) * DD + col] = f2bf(acc[j].f[r] + bvv);
    } else {                              // vT layout [B, D, M]
#pragma unroll
      for (int r = 0; r < 8; ++r)
        out[((size_t)b * DD + col) * M + mb + r] = f2bf(acc[j].f[r] + bvv);
    }
  }
}

// ---------------------- fused attention + max pool -------------------------
// One wave per (b, 16-row q tile). f32 scores + packed-bf16 probs in LDS.
// valid_len (uniform per block) bounds all compute; EXEC stays all-ones.
__global__ __launch_bounds__(32) void k_attn_pool(const uint16_t* __restrict__ qb,
    const uint16_t* __restrict__ kb, const uint16_t* __restrict__ vT,
    const int* __restrict__ valid_len, float* __restrict__ pooled) {
  __shared__ float    sc[16][520];        // raw scores (f32)
  __shared__ uint16_t pbf[16][520];       // unnormalized probs (bf16 pairs)
  const int lane = threadIdx.x;
  const int l = lane & 15, hi = lane >> 4;
  const int tt = blockIdx.x, b = blockIdx.y;
  const int vl = valid_len[b];            // >= 1, uniform across block
  const int ntv = (vl + 15) >> 4;         // score tiles with any valid column
  const int ncv = (vl + 31) >> 5;         // P@V chunks with any valid key

  // ---- phase A: scores = (q @ k^T) * scale, masked, into sc ----
  ABFrag qa[8];
  const uint16_t* qrow = qb + ((size_t)b * TT + tt * 16 + l) * DD;
#pragma unroll
  for (int kc = 0; kc < 8; ++kc) load_a_frag(qa[kc], qrow, kc * 32, hi);

  const float scale = 0.0625f;            // 1/sqrt(256)
  for (int nt = 0; nt < ntv; ++nt) {
    const uint16_t* krow = kb + ((size_t)b * NN + nt * 16 + l) * DD;
    if (nt + 1 < ntv)                     // prefetch next k tile's row
      __builtin_prefetch(krow + 16 * DD, 0, 1);
    CDFrag acc;
#pragma unroll
    for (int i = 0; i < 8; ++i) acc.f[i] = 0.0f;
#pragma unroll
    for (int kc = 0; kc < 8; ++kc) {
      ABFrag bb;
      load_b_frag(bb, krow, kc * 32 + hi * 16);
      acc.v = wmma_bf16(qa[kc], bb, acc.v);
    }
    const int n = nt * 16 + l;
    const bool masked = (n >= vl);
#pragma unroll
    for (int r = 0; r < 8; ++r)
      sc[hi * 8 + r][n] = masked ? -1e9f : acc.f[r] * scale;
  }
  __syncthreads();

  // ---- phase B: softmax stats; store unnormalized exp as packed bf16 ----
  // Lanes (l, l+16) split row l: half hi covers columns [hi*256, hi*256+256).
  const int lo  = hi * 256;
  int cnt = vl - lo; cnt = (cnt < 0) ? 0 : (cnt > 256 ? 256 : cnt);
  float rmax = -3.0e38f;
  for (int j = 0; j < cnt; ++j) rmax = fmaxf(rmax, sc[l][lo + j]);
  rmax = fmaxf(rmax, __shfl_xor(rmax, 16, 32));
  float rsum = 0.0f;
  for (int j = 0; j < 256; j += 2) {
    const float e0 = (j     < cnt) ? __expf(sc[l][lo + j]     - rmax) : 0.0f;
    const float e1 = (j + 1 < cnt) ? __expf(sc[l][lo + j + 1] - rmax) : 0.0f;
    rsum += e0 + e1;
    *(uint32_t*)&pbf[l][lo + j] = (uint32_t)f2bf(e0) | ((uint32_t)f2bf(e1) << 16);
  }
  rsum += __shfl_xor(rsum, 16, 32);
  const float rinv = 1.0f / rsum;         // row l's 1/sum (both half-lanes)
  __syncthreads();

  // ---- phase C: enc = P @ V (row-scaled in epilogue), fused max-pool ----
  // P A-fragments depend only on the K chunk -> build once, reuse for all dt.
  ABFrag pa[16];
#pragma unroll
  for (int nc = 0; nc < 16; ++nc) {
#pragma unroll
    for (int i = 0; i < 8; ++i) {
      const int k = nc * 32 + hi * 16 + 2 * i;
      pa[nc].u[i] = *(const uint32_t*)&pbf[l][k];
    }
  }
  for (int dt = 0; dt < 16; ++dt) {
    const uint16_t* vrow = vT + ((size_t)b * DD + dt * 16 + l) * NN;
    if (dt < 15)                          // prefetch next v tile's row
      __builtin_prefetch(vrow + 16 * NN, 0, 1);
    CDFrag acc;
#pragma unroll
    for (int i = 0; i < 8; ++i) acc.f[i] = 0.0f;
#pragma unroll
    for (int nc = 0; nc < 16; ++nc) {
      if (nc < ncv) {                     // uniform branch: skip all-zero P
        ABFrag bb;
        load_b_frag(bb, vrow, nc * 32 + hi * 16);
        acc.v = wmma_bf16(pa[nc], bb, acc.v);
      }
    }
    // acc.f[r] = enc[row hi*8+r][col dt*16+l]; normalize per row, then max.
    float m = -3.0e38f;
#pragma unroll
    for (int r = 0; r < 8; ++r) {
      const float rr = __shfl(rinv, hi * 8 + r, 32);  // row's 1/sum
      m = fmaxf(m, acc.f[r] * rr);
    }
    m = fmaxf(m, __shfl_xor(m, 16, 32));  // combine rows 0-7 with 8-15
    if (hi == 0) {                        // one lane per output column
      float* addr = pooled + (size_t)b * DD + dt * 16 + l;
      if (m >= 0.0f) atomicMax((int*)addr, __float_as_int(m));
      else           atomicMin((unsigned int*)addr, __float_as_uint(m));
    }
  }
}

// --------------------------- output head -----------------------------------
__global__ __launch_bounds__(256) void k_final(const float* __restrict__ pooled,
    const float* __restrict__ emb, const int* __restrict__ labels,
    const float* __restrict__ w_out, const float* __restrict__ b_out,
    float* __restrict__ out) {
  const int b = blockIdx.x, d = threadIdx.x;
  float v = (pooled[(size_t)b * DD + d] + emb[labels[b] * DD + d]) * w_out[d];
  __shared__ float sh[8];
#pragma unroll
  for (int m = 1; m < 32; m <<= 1) v += __shfl_xor(v, m, 32);
  if ((d & 31) == 0) sh[d >> 5] = v;
  __syncthreads();
  if (d == 0) {
    float t = 0.0f;
    for (int i = 0; i < 8; ++i) t += sh[i];
    out[b] = 1.0f / (1.0f + __expf(-(t + b_out[0])));
  }
}

// --------------------------- launcher ---------------------------------------

extern "C" void kernel_launch(void* const* d_in, const int* in_sizes, int n_in,
                              void* d_out, int out_size, void* d_ws, size_t ws_size,
                              hipStream_t stream) {
  (void)in_sizes; (void)n_in; (void)out_size; (void)ws_size;
  const float* traj      = (const float*)d_in[0];
  const int*   labels    = (const int*)d_in[1];
  const float* h         = (const float*)d_in[2];
  const int*   valid_len = (const int*)d_in[3];
  const float* emb       = (const float*)d_in[4];
  const float* w_mlp     = (const float*)d_in[5];
  const float* b_mlp     = (const float*)d_in[6];
  const float* ln_g      = (const float*)d_in[7];
  const float* ln_b      = (const float*)d_in[8];
  const float* wq        = (const float*)d_in[9];
  const float* bq        = (const float*)d_in[10];
  const float* wk        = (const float*)d_in[11];
  const float* bk        = (const float*)d_in[12];
  const float* wv        = (const float*)d_in[13];
  const float* bv        = (const float*)d_in[14];
  const float* w_out     = (const float*)d_in[15];
  const float* b_out     = (const float*)d_in[16];
  float* out = (float*)d_out;

  char* ws = (char*)d_ws;
  size_t off = 0;
  auto alloc = [&](size_t bytes) -> void* {
    void* p = ws + off;
    off += (bytes + 255) & ~(size_t)255;
    return p;
  };
  uint16_t* xb   = (uint16_t*)alloc((size_t)BB * TT * DD * 2);
  uint16_t* hb   = (uint16_t*)alloc((size_t)BB * NN * DD * 2);
  uint16_t* wqT  = (uint16_t*)alloc((size_t)DD * DD * 2);
  uint16_t* wkT  = (uint16_t*)alloc((size_t)DD * DD * 2);
  uint16_t* wvT  = (uint16_t*)alloc((size_t)DD * DD * 2);
  uint16_t* qb   = (uint16_t*)alloc((size_t)BB * TT * DD * 2);
  uint16_t* kbuf = (uint16_t*)alloc((size_t)BB * NN * DD * 2);
  uint16_t* vTb  = (uint16_t*)alloc((size_t)BB * DD * NN * 2);
  float*    pooled = (float*)alloc((size_t)BB * DD * 4);

  // 1. weight transpose+convert, h convert, fused MLP/LN/embed
  k_transpose_w<<<DD * DD / 256, 256, 0, stream>>>(wq, wqT);
  k_transpose_w<<<DD * DD / 256, 256, 0, stream>>>(wk, wkT);
  k_transpose_w<<<DD * DD / 256, 256, 0, stream>>>(wv, wvT);
  k_convert_bf16<<<(BB * NN * DD) / 256, 256, 0, stream>>>(h, hb, BB * NN * DD);
  k_compute_x<<<dim3(TT, BB), 256, 0, stream>>>(traj, labels, emb, w_mlp, b_mlp,
                                                ln_g, ln_b, xb);
  // 2. projections (WMMA): grid = (M tiles/4, D tiles/4, B)
  k_gemm_proj<<<dim3(TT / 64, DD / 64, BB), 128, 0, stream>>>(xb, wqT, bq, qb, TT, 0);
  k_gemm_proj<<<dim3(NN / 64, DD / 64, BB), 128, 0, stream>>>(hb, wkT, bk, kbuf, NN, 0);
  k_gemm_proj<<<dim3(NN / 64, DD / 64, BB), 128, 0, stream>>>(hb, wvT, bv, vTb, NN, 1);
  // 3. fused attention + max pooling
  k_init_pooled<<<(BB * DD) / 256, 256, 0, stream>>>((uint32_t*)pooled, BB * DD);
  k_attn_pool<<<dim3(TT / 16, BB), 32, 0, stream>>>(qb, kbuf, vTb, valid_len, pooled);
  // 4. head
  k_final<<<BB, 256, 0, stream>>>(pooled, emb, labels, w_out, b_out, out);
}
